// TTTViTBlock_57896159150330
// MI455X (gfx1250) — matrix operand order repaired and checked by reading, not compiled, not run on hardware
//
#include <hip/hip_runtime.h>
#include <math.h>

// ---------------- Problem constants ----------------
#define BB 32
#define LL 1024
#define DD 1024
#define NHH 16
#define II 4096
#define CSS 16
#define NCC 64
#define HDD 64
#define NTOK (BB*LL)         // 32768

// ---------------- gfx1250 async-copy support (guarded) ----------------
#ifdef __has_builtin
#if __has_builtin(__builtin_amdgcn_global_load_async_to_lds_b128)
#define USE_ASYNC_COPY 1
#endif
#if __has_builtin(__builtin_amdgcn_s_wait_asynccnt)
#define HAVE_WAIT_ASYNC_BUILTIN 1
#endif
#endif
#ifndef USE_ASYNC_COPY
#define USE_ASYNC_COPY 0
#endif

// ---------------- Types ----------------
typedef __attribute__((ext_vector_type(16))) __bf16          v16bf;
typedef __attribute__((ext_vector_type(16))) unsigned short  v16us;
typedef __attribute__((ext_vector_type(8)))  float           v8f;
typedef int v4i_vs __attribute__((vector_size(16)));
typedef __attribute__((address_space(1))) v4i_vs* gas_v4;    // global int4*
typedef __attribute__((address_space(3))) v4i_vs* las_v4;    // LDS int4*

// ---------------- Helpers ----------------
__device__ __forceinline__ unsigned short f2bf(float f) {
    unsigned u = __builtin_bit_cast(unsigned, f);
    unsigned r = u + 0x7FFFu + ((u >> 16) & 1u);   // round-to-nearest-even
    return (unsigned short)(r >> 16);
}
__device__ __forceinline__ float bf2f(unsigned short u) {
    unsigned x = ((unsigned)u) << 16;
    return __builtin_bit_cast(float, x);
}

__device__ __forceinline__ v8f wmma_bf16(v16bf a, v16bf b, v8f c) {
    return __builtin_amdgcn_wmma_f32_16x16x32_bf16(
        false, a, false, b, (short)0, c, false, false);
}

// 16-byte global -> LDS copy (async DMA on gfx1250 if available)
__device__ __forceinline__ void cp16(const unsigned short* g, unsigned short* l) {
#if USE_ASYNC_COPY
    __builtin_amdgcn_global_load_async_to_lds_b128((gas_v4)g, (las_v4)l, 0, 0);
#else
    *(uint4*)l = *(const uint4*)g;
#endif
}
template <int N>
__device__ __forceinline__ void wait_async() {
#if USE_ASYNC_COPY
#if defined(HAVE_WAIT_ASYNC_BUILTIN)
    __builtin_amdgcn_s_wait_asynccnt((unsigned short)N);
#else
    asm volatile("s_wait_asynccnt %0" :: "i"(N) : "memory");
#endif
#endif
}

// A-fragment (16x32): lane<16 -> row m=lane, elems K={0..7,16..23};
// lane>=16 -> row m=lane-16, elems K={8..15,24..31}. Source [m][k], k contiguous.
__device__ __forceinline__ v16bf afrag_us(const unsigned short* t, int ld, int lane, int kbase) {
    int half = lane >> 4, m = lane & 15;
    v16us v;
    #pragma unroll
    for (int e = 0; e < 16; ++e) {
        int k = kbase + e + ((e >= 8) ? 8 : 0) + (half ? 8 : 0);
        v[e] = t[m * ld + k];
    }
    return __builtin_bit_cast(v16bf, v);
}
__device__ __forceinline__ v16bf afrag_f32(const float* t, int ld, int lane, int kbase, int kvalid) {
    int half = lane >> 4, m = lane & 15;
    v16us v;
    #pragma unroll
    for (int e = 0; e < 16; ++e) {
        int k = kbase + e + ((e >= 8) ? 8 : 0) + (half ? 8 : 0);
        float x = (k < kvalid) ? t[m * ld + k] : 0.0f;
        v[e] = f2bf(x);
    }
    return __builtin_bit_cast(v16bf, v);
}
// B-fragment (32x16): lane<16 -> col n=lane holds K={0..7,16..23}, lane>=16 -> K={8..15,24..31}.
// Source stored [n][k] with k contiguous.
__device__ __forceinline__ v16bf bfrag_nk_us(const unsigned short* t, int ld, int lane, int kbase) {
    int half = lane >> 4, n = lane & 15;
    v16us v;
    #pragma unroll
    for (int e = 0; e < 16; ++e) {
        int k = kbase + e + ((e >= 8) ? 8 : 0) + (half ? 8 : 0);
        v[e] = t[n * ld + k];
    }
    return __builtin_bit_cast(v16bf, v);
}
__device__ __forceinline__ v16bf bfrag_nk_f32(const float* t, int ld, int lane, int kbase, int kvalid) {
    int half = lane >> 4, n = lane & 15;
    v16us v;
    #pragma unroll
    for (int e = 0; e < 16; ++e) {
        int k = kbase + e + ((e >= 8) ? 8 : 0) + (half ? 8 : 0);
        float x = (k < kvalid) ? t[n * ld + k] : 0.0f;
        v[e] = f2bf(x);
    }
    return __builtin_bit_cast(v16bf, v);
}

// ---------------- Weight conversion (all weights stored B^T = [n][k]) ----------------
__global__ void transpose_bf16_kernel(const float* __restrict__ src, unsigned short* __restrict__ dst,
                                      int R, int C) {
    // src [R][C] row-major -> dst[c*R + r] = bf16(src[r*C + c])
    int i = blockIdx.x * 256 + threadIdx.x;
    if (i >= R * C) return;
    int c = i / R, r = i % R;
    dst[i] = f2bf(src[(size_t)r * C + c]);
}
__global__ void pack_qkvT_kernel(const float* __restrict__ wq, const float* __restrict__ wk,
                                 const float* __restrict__ wv, unsigned short* __restrict__ dst) {
    // dst[c*1024 + r] = bf16(w{q|k|v}[r][c%1024]),  c in [0,3072)
    int i = blockIdx.x * 256 + threadIdx.x;
    if (i >= 3 * DD * DD) return;
    int c = i >> 10, r = i & 1023;
    const float* s = (c < DD) ? wq : ((c < 2 * DD) ? wk : wv);
    dst[i] = f2bf(s[(size_t)r * DD + (c & (DD - 1))]);
}
__global__ void lrw_padT_kernel(const float* __restrict__ lr_w, unsigned short* __restrict__ dst) {
    // lr_w is [16][1024] (already [n][k]); pad to [128][1024]
    int i = blockIdx.x * 256 + threadIdx.x;
    if (i >= 128 * DD) return;
    int n = i >> 10, k = i & 1023;
    dst[i] = (n < NHH) ? f2bf(lr_w[n * DD + k]) : (unsigned short)0;
}

// ---------------- LayerNorm: f32 rows -> bf16 rows ----------------
__global__ __launch_bounds__(256) void ln_bf16_kernel(
    const float* __restrict__ x, const float* __restrict__ w, const float* __restrict__ b,
    unsigned short* __restrict__ out, float eps) {
    __shared__ float sbuf[256], sbuf2[256];
    int row = blockIdx.x;
    const float* xr = x + (size_t)row * DD;
    float s = 0.f, s2 = 0.f;
    for (int i = threadIdx.x; i < DD; i += 256) { float v = xr[i]; s += v; s2 += v * v; }
    sbuf[threadIdx.x] = s; sbuf2[threadIdx.x] = s2;
    __syncthreads();
    for (int off = 128; off > 0; off >>= 1) {
        if (threadIdx.x < off) { sbuf[threadIdx.x] += sbuf[threadIdx.x + off]; sbuf2[threadIdx.x] += sbuf2[threadIdx.x + off]; }
        __syncthreads();
    }
    float mu = sbuf[0] * (1.0f / DD);
    float var = sbuf2[0] * (1.0f / DD) - mu * mu;
    float rstd = rsqrtf(var + eps);
    for (int i = threadIdx.x; i < DD; i += 256)
        out[(size_t)row * DD + i] = f2bf((xr[i] - mu) * rstd * w[i] + b[i]);
}

// ---------------- WMMA GEMM: C = A[M,K] * B[K,N] (+bias, gelu, resid) ----------------
// B supplied TRANSPOSED: Bt[n][k]. Block tile 128x128, K-step 32, 256 threads = 8 waves
// in a 4(M) x 2(N) grid; wave tile 32x64 = 8 WMMA C tiles. Double-buffered LDS stages
// filled with 16B async copies (global_load_async_to_lds_b128) when available.
#define BLK_M 128
#define BLK_N 128
#define BLK_K 32
__global__ __launch_bounds__(256) void gemm_bf16_kernel(
    const unsigned short* __restrict__ A, const unsigned short* __restrict__ Bt,
    int M, int N, int K,
    const float* __restrict__ bias, int act,
    const float* __restrict__ resid,
    float* __restrict__ outF, unsigned short* __restrict__ outBF) {
    __shared__ unsigned short sA[2][BLK_M * BLK_K];
    __shared__ unsigned short sB[2][BLK_N * BLK_K];
    int tid = threadIdx.x, lane = tid & 31, wave = tid >> 5;
    int wm = wave >> 1, wn = wave & 1;
    int m0 = blockIdx.y * BLK_M, n0 = blockIdx.x * BLK_N;
    v8f acc[2][4] = {};

    auto stage = [&](int s, int k0) {
        // 512 16B segments each for A and B; 2+2 per thread
        #pragma unroll
        for (int i = 0; i < 2; ++i) {
            int seg = tid + i * 256;            // 0..511
            int r = seg >> 2, p = seg & 3;      // row, 16B part
            cp16(A  + (size_t)(m0 + r) * K + k0 + p * 8, &sA[s][r * BLK_K + p * 8]);
            cp16(Bt + (size_t)(n0 + r) * K + k0 + p * 8, &sB[s][r * BLK_K + p * 8]);
        }
    };

    int nk = K / BLK_K;
    stage(0, 0);
    for (int ik = 0; ik < nk; ++ik) {
        int s = ik & 1;
        if (ik + 1 < nk) { stage(s ^ 1, (ik + 1) * BLK_K); wait_async<4>(); }
        else             { wait_async<0>(); }
        __syncthreads();
        v16bf af0 = afrag_us(&sA[s][(wm * 32 +  0) * BLK_K], BLK_K, lane, 0);
        v16bf af1 = afrag_us(&sA[s][(wm * 32 + 16) * BLK_K], BLK_K, lane, 0);
        #pragma unroll
        for (int nt = 0; nt < 4; ++nt) {
            v16bf bf = bfrag_nk_us(&sB[s][(wn * 64 + nt * 16) * BLK_K], BLK_K, lane, 0);
            acc[0][nt] = wmma_bf16(af0, bf, acc[0][nt]);
            acc[1][nt] = wmma_bf16(af1, bf, acc[1][nt]);
        }
        __syncthreads();
    }

    int half = lane >> 4, nn = lane & 15;
    #pragma unroll
    for (int i = 0; i < 2; ++i) {
        #pragma unroll
        for (int nt = 0; nt < 4; ++nt) {
            #pragma unroll
            for (int j = 0; j < 8; ++j) {
                int mr = m0 + wm * 32 + i * 16 + j + (half ? 8 : 0);
                int nc = n0 + wn * 64 + nt * 16 + nn;
                float v = acc[i][nt][j];
                if (bias)  v += bias[nc];
                if (act == 1) v = 0.5f * v * (1.0f + erff(v * 0.70710678118654752f));
                if (resid) v += resid[(size_t)mr * N + nc];
                size_t o = (size_t)mr * N + nc;
                if (outF)  outF[o] = v;
                if (outBF) outBF[o] = f2bf(v);
            }
        }
    }
}

// ---------------- lr apply: sigmoid(tmp + lr_b)/HD, scatter to [B][NH][NC][CS] ----------------
__global__ __launch_bounds__(256) void lr_apply_kernel(
    const float* __restrict__ tmp /*[NTOK][128]*/, const float* __restrict__ lr_b,
    float* __restrict__ out) {
    int t = blockIdx.x * 256 + threadIdx.x;
    if (t >= NTOK * NHH) return;
    int token = t >> 4, hh = t & 15;
    float s = tmp[(size_t)token * 128 + hh] + lr_b[hh];
    float val = (1.0f / (1.0f + expf(-s))) * (1.0f / (float)HDD);
    int b = token >> 10, rem = token & 1023, chunk = rem >> 4, cs = rem & 15;
    out[(((size_t)b * NHH + hh) * NCC + chunk) * CSS + cs] = val;
}

// ---------------- TTT scan: one wave per (b, head) ----------------
__global__ __launch_bounds__(32) void ttt_scan_kernel(
    const unsigned short* __restrict__ qkv,   // [32768][3072] bf16 (Q|K|V)
    const float* __restrict__ lr,             // [B][NH][NC][CS]
    const float* __restrict__ W1_init,        // [NH][64][64]
    const float* __restrict__ b1_init,        // [NH][64]
    const float* __restrict__ norm_w,         // [NH][64]
    const float* __restrict__ norm_b,         // [NH][64]
    const float* __restrict__ tok_param,      // [16]
    float* __restrict__ y_inner) {            // [32768][1024]
    __shared__ float sW1T[64 * 64];           // [e][d] = W1[d][e]  (k contiguous for B-frags)
    __shared__ float sb1[64], sGam[64], sBet[64];
    __shared__ float sZ[16 * 64];             // [m][c]
    __shared__ float sGT[64 * 16];            // [c][k] = grad[k][c] (k contiguous)
    __shared__ float sXq[16 * 64], sXk[16 * 64], sXv[16 * 64];
    __shared__ float sS[16 * 16];
    __shared__ float sTok[16], sLr[16], sLE[16];

    int lane = threadIdx.x;
    int bh = blockIdx.x, b = bh >> 4, h = bh & 15;

    for (int i = lane; i < 64 * 64; i += 32) {
        int e = i >> 6, d = i & 63;
        sW1T[i] = W1_init[h * 4096 + d * 64 + e];
    }
    for (int i = lane; i < 64; i += 32) {
        sb1[i]  = b1_init[h * 64 + i];
        sGam[i] = norm_w[h * 64 + i];
        sBet[i] = norm_b[h * 64 + i];
    }
    if (lane < 16) { float t = 1.0f / (float)(lane + 1) + tok_param[lane]; sTok[lane] = t > 0.f ? t : 0.f; }
    __syncthreads();

    int half = lane >> 4, nn = lane & 15;

    for (int n = 0; n < NCC; ++n) {
        for (int i = lane; i < 16 * 64; i += 32) {
            int r = i >> 6, c = i & 63;
            size_t base = (size_t)(b * LL + n * CSS + r) * (3 * DD) + h * HDD + c;
            sXq[i] = bf2f(qkv[base]);
            sXk[i] = bf2f(qkv[base + DD]);
            sXv[i] = bf2f(qkv[base + 2 * DD]);
        }
        if (lane < 16) sLr[lane] = lr[((size_t)bh * NCC + n) * CSS + lane];
        __syncthreads();

        v16bf ak0 = afrag_f32(sXk, 64, lane, 0, 64);
        v16bf ak1 = afrag_f32(sXk, 64, lane, 32, 64);
        v16bf aq0 = afrag_f32(sXq, 64, lane, 0, 64);
        v16bf aq1 = afrag_f32(sXq, 64, lane, 32, 64);

        // ---- Z1 = xk @ W1 + b1 ----
        #pragma unroll
        for (int nt = 0; nt < 4; ++nt) {
            v8f acc = {};
            acc = wmma_bf16(ak0, bfrag_nk_f32(sW1T + nt * 16 * 64, 64, lane, 0, 64), acc);
            acc = wmma_bf16(ak1, bfrag_nk_f32(sW1T + nt * 16 * 64, 64, lane, 32, 64), acc);
            #pragma unroll
            for (int j = 0; j < 8; ++j) {
                int m = j + (half ? 8 : 0);
                sZ[m * 64 + nt * 16 + nn] = acc[j] + sb1[nt * 16 + nn];
            }
        }
        __syncthreads();

        // ---- grad = ln_fused_l2_bwd(Z1, xv - xk, gamma, beta), rowwise; stored transposed ----
        if (lane < 16) {
            const float* z = sZ + lane * 64;
            float mu = 0.f; for (int c = 0; c < 64; ++c) mu += z[c];
            mu *= (1.0f / 64.0f);
            float var = 0.f; for (int c = 0; c < 64; ++c) { float d = z[c] - mu; var += d * d; }
            var *= (1.0f / 64.0f);
            float rstd = rsqrtf(var + 1e-6f);
            float sg = 0.f, sgx = 0.f;
            for (int c = 0; c < 64; ++c) {
                float xh = (z[c] - mu) * rstd;
                float y  = sGam[c] * xh + sBet[c];
                float gx = (y - (sXv[lane * 64 + c] - sXk[lane * 64 + c])) * sGam[c];
                sg += gx; sgx += gx * xh;
            }
            for (int c = 0; c < 64; ++c) {
                float xh = (z[c] - mu) * rstd;
                float y  = sGam[c] * xh + sBet[c];
                float gx = (y - (sXv[lane * 64 + c] - sXk[lane * 64 + c])) * sGam[c];
                sGT[c * 16 + lane] = (64.0f * gx - sg - xh * sgx) * (rstd * (1.0f / 64.0f));
            }
        }
        __syncthreads();

        // ---- Attn = xq @ xk^T ; S = -tril(eta * (1 + Attn)) ----
        {
            v8f attn = {};
            attn = wmma_bf16(aq0, bfrag_nk_f32(sXk, 64, lane, 0, 64),  attn);
            attn = wmma_bf16(aq1, bfrag_nk_f32(sXk, 64, lane, 32, 64), attn);
            #pragma unroll
            for (int j = 0; j < 8; ++j) {
                int q = j + (half ? 8 : 0);
                float v = (q >= nn) ? sTok[q] * sLr[nn] * (1.0f + attn[j]) : 0.0f;
                sS[q * 16 + nn] = -v;
            }
            if (lane < 16) sLE[lane] = sTok[15] * sLr[lane];   // last_eta
        }
        __syncthreads();

        // ---- Z1_bar = xq@W1 + b1 + S@grad ; out = xq + LN(Z1_bar) ----
        v16bf sfr = afrag_f32(sS, 16, lane, 0, 16);
        v16bf gb[4];
        #pragma unroll
        for (int nt = 0; nt < 4; ++nt) gb[nt] = bfrag_nk_f32(sGT + nt * 16 * 16, 16, lane, 0, 16);
        #pragma unroll
        for (int nt = 0; nt < 4; ++nt) {
            v8f acc = {};
            acc = wmma_bf16(aq0, bfrag_nk_f32(sW1T + nt * 16 * 64, 64, lane, 0, 64), acc);
            acc = wmma_bf16(aq1, bfrag_nk_f32(sW1T + nt * 16 * 64, 64, lane, 32, 64), acc);
            acc = wmma_bf16(sfr, gb[nt], acc);
            #pragma unroll
            for (int j = 0; j < 8; ++j) {
                int m = j + (half ? 8 : 0);
                sZ[m * 64 + nt * 16 + nn] = acc[j] + sb1[nt * 16 + nn];
            }
        }
        __syncthreads();

        if (lane < 16) {
            const float* z = sZ + lane * 64;
            float mu = 0.f; for (int c = 0; c < 64; ++c) mu += z[c];
            mu *= (1.0f / 64.0f);
            float var = 0.f; for (int c = 0; c < 64; ++c) { float d = z[c] - mu; var += d * d; }
            var *= (1.0f / 64.0f);
            float rstd = rsqrtf(var + 1e-6f);
            size_t orow = (size_t)(b * LL + n * CSS + lane) * DD + h * HDD;
            for (int c = 0; c < 64; ++c) {
                float xh = (z[c] - mu) * rstd;
                y_inner[orow + c] = sXq[lane * 64 + c] + sGam[c] * xh + sBet[c];
            }
        }

        // ---- W1 -= xk^T diag(last_eta) grad ----
        #pragma unroll
        for (int mt = 0; mt < 4; ++mt) {
            v16us va;
            #pragma unroll
            for (int e = 0; e < 16; ++e) {
                int k = e + ((e >= 8) ? 8 : 0) + (half ? 8 : 0);
                float x = (k < 16) ? (-sLE[k] * sXk[k * 64 + mt * 16 + nn]) : 0.0f;
                va[e] = f2bf(x);
            }
            v16bf am = __builtin_bit_cast(v16bf, va);
            #pragma unroll
            for (int nt = 0; nt < 4; ++nt) {
                v8f c;
                #pragma unroll
                for (int j = 0; j < 8; ++j) {
                    int m = j + (half ? 8 : 0);
                    c[j] = sW1T[(nt * 16 + nn) * 64 + mt * 16 + m];
                }
                c = wmma_bf16(am, gb[nt], c);
                #pragma unroll
                for (int j = 0; j < 8; ++j) {
                    int m = j + (half ? 8 : 0);
                    sW1T[(nt * 16 + nn) * 64 + mt * 16 + m] = c[j];
                }
            }
        }
        // ---- b1 -= sum_k last_eta[k] * grad[k][:] ----
        for (int e = lane; e < 64; e += 32) {
            float s = 0.f;
            for (int k = 0; k < 16; ++k) s += sLE[k] * sGT[e * 16 + k];
            sb1[e] -= s;
        }
        __syncthreads();
    }
}

// ---------------- Host-side launch ----------------
extern "C" void kernel_launch(void* const* d_in, const int* in_sizes, int n_in,
                              void* d_out, int out_size, void* d_ws, size_t ws_size,
                              hipStream_t stream) {
    const float* x        = (const float*)d_in[0];
    const float* ln1_w    = (const float*)d_in[2];
    const float* ln1_b    = (const float*)d_in[3];
    const float* wq       = (const float*)d_in[4];
    const float* wk       = (const float*)d_in[5];
    const float* wv       = (const float*)d_in[6];
    const float* wo       = (const float*)d_in[7];
    const float* W1_init  = (const float*)d_in[8];
    const float* b1_init  = (const float*)d_in[9];
    const float* tnw      = (const float*)d_in[10];
    const float* tnb      = (const float*)d_in[11];
    const float* lr_w     = (const float*)d_in[12];
    const float* lr_b     = (const float*)d_in[13];
    const float* tokp     = (const float*)d_in[14];
    const float* pnw      = (const float*)d_in[15];
    const float* pnb      = (const float*)d_in[16];
    const float* ln2_w    = (const float*)d_in[17];
    const float* ln2_b    = (const float*)d_in[18];
    const float* w_in     = (const float*)d_in[19];
    const float* b_in     = (const float*)d_in[20];
    const float* w_out    = (const float*)d_in[21];
    const float* b_out    = (const float*)d_in[22];
    float* out = (float*)d_out;

    char* ws = (char*)d_ws;
    size_t off = 0;
    auto alloc = [&](size_t bytes) { size_t o = off; off += (bytes + 255) & ~(size_t)255; return o; };
    unsigned short* h_bf    = (unsigned short*)(ws + alloc((size_t)NTOK * DD * 2));
    unsigned short* wcatT   = (unsigned short*)(ws + alloc((size_t)DD * 3 * DD * 2));
    unsigned short* woT     = (unsigned short*)(ws + alloc((size_t)DD * DD * 2));
    unsigned short* winT    = (unsigned short*)(ws + alloc((size_t)DD * II * 2));
    unsigned short* woutT   = (unsigned short*)(ws + alloc((size_t)II * DD * 2));
    unsigned short* lrwT    = (unsigned short*)(ws + alloc((size_t)128 * DD * 2));
    unsigned short* qkv_bf  = (unsigned short*)(ws + alloc((size_t)NTOK * 3 * DD * 2));
    float*          lrtmp   = (float*)(ws + alloc((size_t)NTOK * 128 * 4));
    float*          lr_f    = (float*)(ws + alloc((size_t)NTOK * NHH * 4));
    float*          yin_f   = (float*)(ws + alloc((size_t)NTOK * DD * 4));
    unsigned short* yp_bf   = (unsigned short*)(ws + alloc((size_t)NTOK * DD * 2));
    float*          y_f     = (float*)(ws + alloc((size_t)NTOK * DD * 4));
    unsigned short* h2_bf   = (unsigned short*)(ws + alloc((size_t)NTOK * DD * 2));
    unsigned short* g_bf    = (unsigned short*)(ws + alloc((size_t)NTOK * II * 2));

    // 1. weight conversion: everything stored transposed ([n][k]) in bf16
    pack_qkvT_kernel<<<(3 * DD * DD + 255) / 256, 256, 0, stream>>>(wq, wk, wv, wcatT);
    transpose_bf16_kernel<<<(DD * DD + 255) / 256, 256, 0, stream>>>(wo, woT, DD, DD);
    transpose_bf16_kernel<<<(DD * II + 255) / 256, 256, 0, stream>>>(w_in, winT, DD, II);
    transpose_bf16_kernel<<<(II * DD + 255) / 256, 256, 0, stream>>>(w_out, woutT, II, DD);
    lrw_padT_kernel<<<(128 * DD + 255) / 256, 256, 0, stream>>>(lr_w, lrwT);

    // 2. h = LN(x) -> bf16
    ln_bf16_kernel<<<NTOK, 256, 0, stream>>>(x, ln1_w, ln1_b, h_bf, 1e-5f);
    // 3. QKV = h @ [wq|wk|wv]
    gemm_bf16_kernel<<<dim3(3 * DD / BLK_N, NTOK / BLK_M), 256, 0, stream>>>(
        h_bf, wcatT, NTOK, 3 * DD, DD, nullptr, 0, nullptr, nullptr, qkv_bf);
    // 4. ttt_lr: WMMA GEMM against padded lr weights, then sigmoid-scatter
    gemm_bf16_kernel<<<dim3(1, NTOK / BLK_M), 256, 0, stream>>>(
        h_bf, lrwT, NTOK, 128, DD, nullptr, 0, nullptr, lrtmp, nullptr);
    lr_apply_kernel<<<(NTOK * NHH + 255) / 256, 256, 0, stream>>>(lrtmp, lr_b, lr_f);
    // 5. TTT scan
    ttt_scan_kernel<<<BB * NHH, 32, 0, stream>>>(qkv_bf, lr_f, W1_init, b1_init, tnw, tnb, tokp, yin_f);
    // 6. post-norm -> bf16
    ln_bf16_kernel<<<NTOK, 256, 0, stream>>>(yin_f, pnw, pnb, yp_bf, 1e-5f);
    // 7. y = x + yp @ wo
    gemm_bf16_kernel<<<dim3(DD / BLK_N, NTOK / BLK_M), 256, 0, stream>>>(
        yp_bf, woT, NTOK, DD, DD, nullptr, 0, x, y_f, nullptr);
    // 8. h2 = LN(y)
    ln_bf16_kernel<<<NTOK, 256, 0, stream>>>(y_f, ln2_w, ln2_b, h2_bf, 1e-5f);
    // 9. g = gelu(h2 @ w_in + b_in)
    gemm_bf16_kernel<<<dim3(II / BLK_N, NTOK / BLK_M), 256, 0, stream>>>(
        h2_bf, winT, NTOK, II, DD, b_in, 1, nullptr, nullptr, g_bf);
    // 10. out = y + g @ w_out + b_out
    gemm_bf16_kernel<<<dim3(DD / BLK_N, NTOK / BLK_M), 256, 0, stream>>>(
        g_bf, woutT, NTOK, DD, II, b_out, 0, y_f, out, nullptr);
}
